// Mambas_82635170775465
// MI455X (gfx1250) — matrix operand order repaired and checked by reading, not compiled, not run on hardware
//
#include <hip/hip_runtime.h>
#include <cstddef>
#include <cstdint>

// ---------------------------------------------------------------------------
// Mamba (DEPTH=6) forward for MI455X / gfx1250, wave32.
//
// Roofline: ~165 GFLOP vs ~1 GB of traffic -> memory bound at 23.3 TB/s.
// Strategy:
//   * all GEMM operands pre-converted to bf16 once (halves GEMM read traffic,
//     removes cvt VALU work from the K-loop),
//   * GEMM tiles staged to LDS by the Tensor Data Mover, DOUBLE BUFFERED:
//     TDM streams K-tile k+1 while the waves run WMMA on K-tile k
//     (TDM ops complete in-order per wave -> s_wait_tensorcnt 2 pipelining),
//   * v_wmma_f32_16x16x32_bf16 (8 per K-step per wave, A-frags reused),
//   * fused epilogues (softplus+bias, dual fp32/bf16 stores),
//   * register-resident sequential selective scan (h[16] in VGPRs).
// Workspace ~210 MB (mostly resident in the 192 MB global L2).
// ---------------------------------------------------------------------------

#define DIM      512
#define DEPTH    6
#define BATCH    4
#define SEQ      2048
#define D_INNER  1024
#define D_STATE  16
#define D_CONV   4
#define DT_RANK  32
#define MROWS    (BATCH * SEQ)          // 8192 rows of activations

typedef __attribute__((ext_vector_type(16))) __bf16        v16bf;
typedef __attribute__((ext_vector_type(8)))  float         v8f;
typedef __attribute__((ext_vector_type(4)))  unsigned int  v4u;
typedef __attribute__((ext_vector_type(8)))  int           v8i_t;
typedef __attribute__((ext_vector_type(4)))  int           v4i_t;

union FragBF {                          // one WMMA A/B operand (8 VGPRs)
    v16bf v;
    uint4 q[2];
};

__device__ __forceinline__ unsigned short f2bf(float f) {
    // fp32 -> bf16 round-to-nearest-even
    unsigned int u = __float_as_uint(f);
    unsigned int r = u + 0x7FFFu + ((u >> 16) & 1u);
    return (unsigned short)(r >> 16);
}

__device__ __forceinline__ float siluf(float x) {
    return x / (1.f + __expf(-x));
}

#if __has_builtin(__builtin_amdgcn_tensor_load_to_lds)
#define HAVE_TDM 1
#else
#define HAVE_TDM 0
#endif

#if HAVE_TDM
// ---------------------------------------------------------------------------
// Issue one TDM 2-D tile load: rows x 32 bf16 elements, row stride
// `stride_elems`, into LDS at byte offset `lds_off`.  D# per ISA ch.8:
//   group0: count=1 | lds_addr | global_addr[56:0] | type=2
//   group1: data_size=1(2B), tensor_dim0=32, tensor_dim1=rows,
//           tile_dim0=32, tile_dim1=rows, tensor_dim0_stride=stride_elems
// workgroup_mask=0 (not in a cluster).  TENSORcnt tracks completion,
// in-order per wave -> usable for software pipelining.
// ---------------------------------------------------------------------------
__device__ __forceinline__ void tdm_load_tile_bf16(unsigned lds_off,
                                                   const void* gsrc,
                                                   unsigned rows,
                                                   unsigned stride_elems)
{
    unsigned long long ga = (unsigned long long)(uintptr_t)gsrc;
    v4u g0;
    g0[0] = 1u;                                        // count=1 (valid, user)
    g0[1] = lds_off;                                   // lds_addr (bytes)
    g0[2] = (unsigned)ga;                              // global_addr[31:0]
    g0[3] = (unsigned)((ga >> 32) & 0x1FFFFFFu) | (2u << 30);  // [56:32]|type=2
    v8i_t g1;
    g1[0] = (int)(1u << 16);                           // data_size = 2 bytes
    g1[1] = (int)(32u << 16);                          // tensor_dim0 = 32
    g1[2] = (int)(rows << 16);                         // tensor_dim1 = rows
    g1[3] = (int)(32u << 16);                          // tile_dim0 = 32
    g1[4] = (int)rows;                                 // tile_dim1 = rows
    g1[5] = (int)stride_elems;                         // tensor_dim0_stride lo
    g1[6] = 0;                                         // stride hi | dim1_stride
    g1[7] = 0;
    v4i_t g2 = {0, 0, 0, 0};
    v4i_t g3 = {0, 0, 0, 0};
#if defined(__clang_major__) && (__clang_major__ >= 23)
    v8i_t g4 = {0, 0, 0, 0, 0, 0, 0, 0};
    __builtin_amdgcn_tensor_load_to_lds(g0, g1, g2, g3, g4, 0);
#else
    __builtin_amdgcn_tensor_load_to_lds(g0, g1, g2, g3, 0);
#endif
}
#endif // HAVE_TDM

// ---------------------------------------------------------------------------
// Tiled GEMM:  C[M,N] = A[M,K] (bf16, row major, stride lda) x W[N,K]^T (bf16)
// BK=32 (one v_wmma_f32_16x16x32_bf16 K-step). (BM/WM)*(BN/WN) waves/block,
// each wave owns a WMxWN tile.  LDS tiles double-buffered: TDM streams tile
// k+1 while waves compute on tile k.
// EPI: 0 = f32 store, 1 = bias+softplus f32 store, 2 = f32 + bf16 dual store.
// ---------------------------------------------------------------------------
template <int BM, int BN, int WM, int WN, int EPI>
__global__ __launch_bounds__((BM / WM) * (BN / WN) * 32)
void gemm_bf16_wmma(const unsigned short* __restrict__ A, int lda,
                    const unsigned short* __restrict__ W,
                    const float* __restrict__ bias,
                    float* __restrict__ C, int ldc,
                    unsigned short* __restrict__ Cb, int K)
{
    constexpr int BK    = 32;
    constexpr int NW    = (BM / WM) * (BN / WN);
    constexpr int NT    = NW * 32;
    constexpr int TM    = WM / 16;
    constexpr int TN    = WN / 16;
    constexpr int WCOLS = BN / WN;

    __shared__ __align__(16) unsigned short As[2][BM * BK];
    __shared__ __align__(16) unsigned short Bs[2][BN * BK];

    const int tid  = threadIdx.x;
    const int m0   = blockIdx.x * BM;
    const int n0   = blockIdx.y * BN;
    const int wid  = tid >> 5;
    const int lane = tid & 31;
    const int half = lane >> 4;       // wave32 half (ISA 7.12.2 layouts)
    const int mrow = lane & 15;
    const int wr   = wid / WCOLS;
    const int wc   = wid % WCOLS;

    const int nk = K / BK;

    v8f acc[TM][TN] = {};

#if HAVE_TDM
    // prologue: stream K-tile 0 into buffer 0
    if (wid == 0) {
        tdm_load_tile_bf16((unsigned)(uintptr_t)&As[0][0],
                           A + (size_t)m0 * lda, BM, (unsigned)lda);
        tdm_load_tile_bf16((unsigned)(uintptr_t)&Bs[0][0],
                           W + (size_t)n0 * K, BN, (unsigned)K);
    }
#endif

    for (int kk = 0; kk < nk; ++kk) {
#if HAVE_TDM
        const int cur = kk & 1;
        const int nxt = cur ^ 1;
        __syncthreads();              // readers of buf `nxt` (iter kk-1) done
        if (wid == 0) {
            if (kk + 1 < nk) {
                // stream next K-tile while this one is computed on
                int k1 = (kk + 1) * BK;
                tdm_load_tile_bf16((unsigned)(uintptr_t)&As[nxt][0],
                                   A + (size_t)m0 * lda + k1, BM, (unsigned)lda);
                tdm_load_tile_bf16((unsigned)(uintptr_t)&Bs[nxt][0],
                                   W + (size_t)n0 * K + k1, BN, (unsigned)K);
                // in-order completion: cnt<=2 => buf `cur` pair has landed
                __builtin_amdgcn_s_wait_tensorcnt(2);
            } else {
                __builtin_amdgcn_s_wait_tensorcnt(0);
            }
        }
        __syncthreads();              // buf `cur` visible to all waves
#else
        const int cur = 0;
        const int k0  = kk * BK;
        __syncthreads();
        for (int i = tid; i < BM * 4; i += NT) {
            int row = i >> 2, c = i & 3;
            *(uint4*)&As[0][row * BK + c * 8] =
                *(const uint4*)(A + (size_t)(m0 + row) * lda + k0 + c * 8);
        }
        for (int i = tid; i < BN * 4; i += NT) {
            int row = i >> 2, c = i & 3;
            *(uint4*)&Bs[0][row * BK + c * 8] =
                *(const uint4*)(W + (size_t)(n0 + row) * K + k0 + c * 8);
        }
        __syncthreads();
#endif

        // ---- fragments from LDS (ds_load_b128 pairs) ----
        const unsigned short* Asb = &As[cur][0];
        const unsigned short* Bsb = &Bs[cur][0];
        FragBF a[TM], b[TN];
#pragma unroll
        for (int tm = 0; tm < TM; ++tm) {
            int r = (wr * WM + tm * 16 + mrow) * BK;
            // A 16-bit 16x32 layout: per lane K in [8h,8h+8) u [16+8h,16+8h+8)
            a[tm].q[0] = *(const uint4*)&Asb[r + half * 8];
            a[tm].q[1] = *(const uint4*)&Asb[r + 16 + half * 8];
        }
#pragma unroll
        for (int tn = 0; tn < TN; ++tn) {
            int r = (wc * WN + tn * 16 + mrow) * BK;
            // B 16-bit 32x16 layout: per lane K in [8h,8h+8) u [16+8h,16+8h+8)
            b[tn].q[0] = *(const uint4*)&Bsb[r + half * 8];
            b[tn].q[1] = *(const uint4*)&Bsb[r + 16 + half * 8];
        }
#pragma unroll
        for (int tm = 0; tm < TM; ++tm)
#pragma unroll
            for (int tn = 0; tn < TN; ++tn)
                acc[tm][tn] = __builtin_amdgcn_wmma_f32_16x16x32_bf16(
                    false, a[tm].v, false, b[tn].v, (short)0, acc[tm][tn],
                    false, false);
    }

    // ---- epilogue / store (C 16x16 f32 layout: VGPR g -> M = g + 8*half) ----
#pragma unroll
    for (int tm = 0; tm < TM; ++tm) {
#pragma unroll
        for (int tn = 0; tn < TN; ++tn) {
            int gm = m0 + wr * WM + tm * 16 + half * 8;
            int gn = n0 + wc * WN + tn * 16 + mrow;
#pragma unroll
            for (int g = 0; g < 8; ++g) {
                float v = acc[tm][tn][g];
                if (EPI == 1) {                      // dt: bias + softplus
                    v += bias[gn];
                    v = (v > 20.f) ? v : log1pf(__expf(v));
                }
                C[(size_t)(gm + g) * ldc + gn] = v;
                if (EPI == 2)                        // bf16 shadow for next GEMM
                    Cb[(size_t)(gm + g) * ldc + gn] = f2bf(v);
            }
        }
    }
}

// ---------------------------------------------------------------------------
// fp32 -> bf16 bulk convert (float4 granularity; n4 = n/4)
// ---------------------------------------------------------------------------
__global__ __launch_bounds__(256)
void f32_to_bf16_kernel(const float* __restrict__ in,
                        unsigned short* __restrict__ out, int n4)
{
    int i = blockIdx.x * 256 + threadIdx.x;
    if (i < n4) {
        float4 v = ((const float4*)in)[i];
        ushort4 s;
        s.x = f2bf(v.x); s.y = f2bf(v.y); s.z = f2bf(v.z); s.w = f2bf(v.w);
        ((ushort4*)out)[i] = s;
    }
}

// ---------------------------------------------------------------------------
// Depthwise causal conv (K=4) + bias + SiLU. Reads xi = xz[:, :, 0:1024].
// Writes fp32 (scan input) and bf16 (x-proj GEMM A operand).
// ---------------------------------------------------------------------------
__global__ __launch_bounds__(256)
void conv_silu_kernel(const float* __restrict__ xz,
                      const float* __restrict__ cw,
                      const float* __restrict__ cb,
                      float* __restrict__ xic,
                      unsigned short* __restrict__ xicb)
{
    int idx = blockIdx.x * blockDim.x + threadIdx.x;   // over MROWS*D_INNER
    int d   = idx & (D_INNER - 1);
    int rl  = idx >> 10;                               // b*SEQ + l
    int l   = rl & (SEQ - 1);

    float acc = cb[d];
#pragma unroll
    for (int j = 0; j < D_CONV; ++j) {
        int t = l - (D_CONV - 1) + j;
        if (t >= 0)
            acc += xz[(size_t)(rl - (D_CONV - 1) + j) * (2 * D_INNER) + d] *
                   cw[d * D_CONV + j];
    }
    acc = siluf(acc);
    xic[(size_t)rl * D_INNER + d]  = acc;
    xicb[(size_t)rl * D_INNER + d] = f2bf(acc);
}

// ---------------------------------------------------------------------------
// Selective scan: one thread per (batch, channel); h[16] lives in VGPRs.
// Fuses skip term (D*u) and the SiLU(z) gate; emits bf16 (out-proj A operand).
// ---------------------------------------------------------------------------
__global__ __launch_bounds__(1024)
void scan_kernel(const float* __restrict__ dt,
                 const float* __restrict__ xdbl,
                 const float* __restrict__ xic,
                 const float* __restrict__ xz,
                 const float* __restrict__ A_log,
                 const float* __restrict__ Dp,
                 unsigned short* __restrict__ yb)
{
    const int b = blockIdx.x;
    const int d = threadIdx.x;

    float A[D_STATE], h[D_STATE];
#pragma unroll
    for (int n = 0; n < D_STATE; ++n) {
        A[n] = -__expf(A_log[d * D_STATE + n]);
        h[n] = 0.f;
    }
    const float Dv = Dp[d];

    for (int t = 0; t < SEQ; ++t) {
        size_t r   = (size_t)b * SEQ + t;
        float dtv  = dt[r * D_INNER + d];
        float xv   = xic[r * D_INNER + d];
        const float* p = xdbl + r * (DT_RANK + 2 * D_STATE);   // 64 floats/row
        float dtx  = dtv * xv;
        float yv   = 0.f;
#pragma unroll
        for (int n = 0; n < D_STATE; ++n) {
            float Bt = p[DT_RANK + n];
            float Ct = p[DT_RANK + D_STATE + n];
            float dA = __expf(dtv * A[n]);
            h[n] = dA * h[n] + dtx * Bt;
            yv  += h[n] * Ct;
        }
        float zv = xz[r * (2 * D_INNER) + D_INNER + d];
        yb[r * D_INNER + d] = f2bf((yv + xv * Dv) * siluf(zv));
    }
}

// ---------------------------------------------------------------------------
// In-place LayerNorm over D=512 (one block / row, 2 elements / thread),
// optional bf16 shadow copy (next layer's GEMM A operand).
// ---------------------------------------------------------------------------
__global__ __launch_bounds__(256)
void ln_kernel(float* __restrict__ x,
               const float* __restrict__ g,
               const float* __restrict__ b,
               unsigned short* __restrict__ xb)
{
    const int row = blockIdx.x;
    const int tid = threadIdx.x;
    __shared__ float red[256];

    float v0 = x[(size_t)row * DIM + tid];
    float v1 = x[(size_t)row * DIM + tid + 256];

    red[tid] = v0 + v1;
    __syncthreads();
    for (int o = 128; o > 0; o >>= 1) {
        if (tid < o) red[tid] += red[tid + o];
        __syncthreads();
    }
    float mu = red[0] * (1.f / DIM);
    __syncthreads();

    float d0 = v0 - mu, d1 = v1 - mu;
    red[tid] = d0 * d0 + d1 * d1;
    __syncthreads();
    for (int o = 128; o > 0; o >>= 1) {
        if (tid < o) red[tid] += red[tid + o];
        __syncthreads();
    }
    float rstd = rsqrtf(red[0] * (1.f / DIM) + 1e-5f);

    float o0 = d0 * rstd * g[tid] + b[tid];
    float o1 = d1 * rstd * g[tid + 256] + b[tid + 256];
    x[(size_t)row * DIM + tid]       = o0;
    x[(size_t)row * DIM + tid + 256] = o1;
    if (xb) {
        xb[(size_t)row * DIM + tid]       = f2bf(o0);
        xb[(size_t)row * DIM + tid + 256] = f2bf(o1);
    }
}

// ---------------------------------------------------------------------------
// Host orchestration: bf16 pre-conversion + 7 kernels/layer, all on `stream`.
// ---------------------------------------------------------------------------
extern "C" void kernel_launch(void* const* d_in, const int* in_sizes, int n_in,
                              void* d_out, int out_size, void* d_ws, size_t ws_size,
                              hipStream_t stream)
{
    (void)in_sizes; (void)n_in; (void)out_size; (void)ws_size;

    const float* X      = (const float*)d_in[0];
    const float* Win    = (const float*)d_in[1];
    const float* conv_w = (const float*)d_in[2];
    const float* conv_b = (const float*)d_in[3];
    const float* Wx     = (const float*)d_in[4];
    const float* Wdt    = (const float*)d_in[5];
    const float* bdt    = (const float*)d_in[6];
    const float* A_log  = (const float*)d_in[7];
    const float* Dp     = (const float*)d_in[8];
    const float* Wout   = (const float*)d_in[9];
    const float* ln_g   = (const float*)d_in[10];
    const float* ln_b   = (const float*)d_in[11];

    // ---- workspace carve-out ----
    char* p = (char*)d_ws;
    auto alloc_f = [&](size_t n) { float* r = (float*)p; p += n * 4; return r; };
    auto alloc_h = [&](size_t n) { unsigned short* r = (unsigned short*)p; p += n * 2; return r; };

    float* xz    = alloc_f((size_t)MROWS * 2 * D_INNER);   // 64 MB
    float* xic   = alloc_f((size_t)MROWS * D_INNER);       // 32 MB
    float* xdbl  = alloc_f((size_t)MROWS * 64);            //  2 MB
    float* dtb   = alloc_f((size_t)MROWS * D_INNER);       // 32 MB
    float* xbuf  = alloc_f((size_t)MROWS * DIM);           // 16 MB
    unsigned short* xabf  = alloc_h((size_t)MROWS * DIM);      //  8 MB
    unsigned short* xicb  = alloc_h((size_t)MROWS * D_INNER);  // 16 MB
    unsigned short* xdblb = alloc_h((size_t)MROWS * 64);       //  1 MB
    unsigned short* ybf   = alloc_h((size_t)MROWS * D_INNER);  // 16 MB
    unsigned short* Winb  = alloc_h((size_t)DEPTH * 2 * D_INNER * DIM);
    unsigned short* Wxb   = alloc_h((size_t)DEPTH * 64 * D_INNER);
    unsigned short* Wdtb  = alloc_h((size_t)DEPTH * D_INNER * DT_RANK);
    unsigned short* Woutb = alloc_h((size_t)DEPTH * DIM * D_INNER);

    // ---- one-time bf16 conversions (cheap, memory bound) ----
    auto cvt = [&](const float* src, unsigned short* dst, size_t n) {
        int n4 = (int)(n / 4);
        f32_to_bf16_kernel<<<(n4 + 255) / 256, 256, 0, stream>>>(src, dst, n4);
    };
    cvt(X,    xabf,  (size_t)MROWS * DIM);
    cvt(Win,  Winb,  (size_t)DEPTH * 2 * D_INNER * DIM);
    cvt(Wx,   Wxb,   (size_t)DEPTH * 64 * D_INNER);
    cvt(Wdt,  Wdtb,  (size_t)DEPTH * D_INNER * DT_RANK);
    cvt(Wout, Woutb, (size_t)DEPTH * DIM * D_INNER);

    for (int l = 0; l < DEPTH; ++l) {
        const unsigned short* Winl  = Winb  + (size_t)l * 2 * D_INNER * DIM;
        const unsigned short* Wxl   = Wxb   + (size_t)l * 64 * D_INNER;
        const unsigned short* Wdtl  = Wdtb  + (size_t)l * D_INNER * DT_RANK;
        const unsigned short* Woutl = Woutb + (size_t)l * DIM * D_INNER;
        const float* cwl  = conv_w + (size_t)l * D_INNER * D_CONV;
        const float* cbl  = conv_b + (size_t)l * D_INNER;
        const float* bdtl = bdt    + (size_t)l * D_INNER;
        const float* Algl = A_log  + (size_t)l * D_INNER * D_STATE;
        const float* Dpl  = Dp     + (size_t)l * D_INNER;
        const float* lngl = ln_g   + (size_t)l * DIM;
        const float* lnbl = ln_b   + (size_t)l * DIM;

        float* x_out = (l == DEPTH - 1) ? (float*)d_out : xbuf;
        unsigned short* xb_next = (l == DEPTH - 1) ? nullptr : xabf;

        // 1) xz = x @ Win^T    (8192 x 2048, K=512), fp32 out
        gemm_bf16_wmma<128, 128, 32, 64, 0>
            <<<dim3(MROWS / 128, (2 * D_INNER) / 128), 256, 0, stream>>>(
                xabf, DIM, Winl, nullptr, xz, 2 * D_INNER, nullptr, DIM);

        // 2) depthwise causal conv + SiLU  (fp32 + bf16 out)
        conv_silu_kernel<<<(MROWS * D_INNER) / 256, 256, 0, stream>>>(
            xz, cwl, cbl, xic, xicb);

        // 3) x_dbl = xi @ Wx^T (8192 x 64, K=1024), fp32 + bf16 out
        gemm_bf16_wmma<128, 64, 32, 32, 2>
            <<<dim3(MROWS / 128, 1), 256, 0, stream>>>(
                xicb, D_INNER, Wxl, nullptr, xdbl, 64, xdblb, D_INNER);

        // 4) dt = softplus(x_dbl[:, :32] @ Wdt^T + bdt)  (8192 x 1024, K=32)
        gemm_bf16_wmma<128, 128, 32, 64, 1>
            <<<dim3(MROWS / 128, D_INNER / 128), 256, 0, stream>>>(
                xdblb, 64, Wdtl, bdtl, dtb, D_INNER, nullptr, DT_RANK);

        // 5) selective scan (+ D*u skip + SiLU gate), bf16 out
        scan_kernel<<<BATCH, D_INNER, 0, stream>>>(
            dtb, xdbl, xic, xz, Algl, Dpl, ybf);

        // 6) out = y @ Wout^T  (8192 x 512, K=1024), fp32 out
        gemm_bf16_wmma<128, 128, 32, 64, 0>
            <<<dim3(MROWS / 128, DIM / 128), 256, 0, stream>>>(
                ybf, D_INNER, Woutl, nullptr, x_out, DIM, nullptr, D_INNER);

        // 7) LayerNorm (in place) + bf16 shadow for next layer
        ln_kernel<<<MROWS, 256, 0, stream>>>(x_out, lngl, lnbl, xb_next);
    }
}
